// SNN_20933670601082
// MI455X (gfx1250) — matrix operand order repaired and checked by reading, not compiled, not run on hardware
//
#include <hip/hip_runtime.h>

// ---------------------------------------------------------------------------
// EventProp LIF SNN forward on MI455X (gfx1250, wave32, WMMA).
// Pipeline: cvt(f32->f16, pad K) -> GEMM1 (WMMA f16->f32) -> scan1 (LIF) ->
//           GEMM2 (WMMA) -> scan2 (LIF, writes spikes to d_out).
// GEMM: 128x128 block tile, K-chunks of 32, double-buffered LDS, staged with
// gfx1250 GLOBAL_LOAD_ASYNC_TO_LDS_B128 (ASYNCcnt) when available.
// ---------------------------------------------------------------------------

typedef _Float16 v8h  __attribute__((ext_vector_type(8)));
typedef _Float16 v16h __attribute__((ext_vector_type(16)));
typedef float    v8f  __attribute__((ext_vector_type(8)));

#define T_STEPS 100
#define BATCH   256
#define D0      784
#define D1      1024
#define D2      256
#define K1PAD   800          // 784 padded to a multiple of 32
#define ALPHA_F 0.8f         // 1 - dt/tau_s
#define BETA_F  0.95f        // 1 - dt/tau_m

#if __has_builtin(__builtin_amdgcn_global_load_async_to_lds_b128)
#define ASYNC_LDS 1
// Builtin signature (from clang diagnostic): param 0 is a pointer to a
// GCC-vector 'int __attribute__((vector_size(16)))' in the global address
// space; param 1 is the LDS-side pointer; then imm offset and cpol.
typedef int gcc_v4i __attribute__((vector_size(16)));
typedef __attribute__((address_space(1))) gcc_v4i glob_v4i;
typedef __attribute__((address_space(3))) gcc_v4i lds_v4i;
#define ASYNC_CP_B128(gp, lp) \
  __builtin_amdgcn_global_load_async_to_lds_b128((glob_v4i*)(gp), (lds_v4i*)(lp), 0, 0)
#else
#define ASYNC_LDS 0
#endif

static __device__ __forceinline__ void wait_async_zero() {
#if ASYNC_LDS
#if __has_builtin(__builtin_amdgcn_s_wait_asynccnt)
  __builtin_amdgcn_s_wait_asynccnt(0);
#else
  asm volatile("s_wait_asynccnt 0x0" ::: "memory");
#endif
#endif
}

// ---------------------------------------------------------------------------
// f32 -> f16 convert with zero-padding of the K dimension.
// ---------------------------------------------------------------------------
__global__ void cvt_pad_kernel(const float* __restrict__ in, _Float16* __restrict__ out,
                               int rows, int kin, int kout) {
  int idx = blockIdx.x * blockDim.x + threadIdx.x;
  if (idx >= rows * kout) return;
  int r = idx / kout;
  int k = idx - r * kout;
  float v = (k < kin) ? in[(size_t)r * kin + k] : 0.0f;
  out[idx] = (_Float16)v;
}

// ---------------------------------------------------------------------------
// C[M,N] = A[M,K] * Bw[N,K]^T   (A,Bw f16 row-major; C f32; f32 accumulate)
// 256 threads = 8 waves. Wave tile 32(M) x 64(N) = 2x4 WMMA accumulators.
// ---------------------------------------------------------------------------
#define LDSS 40                        // 32 + 8 halves pad per LDS row
__global__ __launch_bounds__(256) void gemm_wmma_f16(
    const _Float16* __restrict__ A, const _Float16* __restrict__ Bw,
    float* __restrict__ C, int M, int N, int K) {
  __shared__ _Float16 As[2][128 * LDSS];
  __shared__ _Float16 Bs[2][128 * LDSS];

  const int m0   = blockIdx.x * 128;
  const int n0   = blockIdx.y * 128;
  const int tid  = threadIdx.x;
  const int wave = tid >> 5;
  const int lane = tid & 31;
  const int wm   = wave & 3;           // 4 wave rows  -> 128 M
  const int wn   = wave >> 2;          // 2 wave cols  -> 128 N (64 each)
  const int half_id = lane >> 4;
  const int lrow    = lane & 15;

  // ---- per-thread staging slots: rows (tid>>2) and (tid>>2)+64, col (tid&3)*8
  const int srow = tid >> 2;
  const int scol = (tid & 3) << 3;
  const _Float16* gA0 = A  + (size_t)(m0 + srow)      * K + scol;
  const _Float16* gA1 = A  + (size_t)(m0 + srow + 64) * K + scol;
  const _Float16* gB0 = Bw + (size_t)(n0 + srow)      * K + scol;
  const _Float16* gB1 = Bw + (size_t)(n0 + srow + 64) * K + scol;
  const int so0 = srow * LDSS + scol;
  const int so1 = (srow + 64) * LDSS + scol;

  v8f acc[2][4];
  const v8f vzero = {0.f,0.f,0.f,0.f,0.f,0.f,0.f,0.f};
#pragma unroll
  for (int mi = 0; mi < 2; ++mi)
#pragma unroll
    for (int ni = 0; ni < 4; ++ni) acc[mi][ni] = vzero;

  const int nk = K >> 5;               // K / 32 chunks

#if ASYNC_LDS
  // ---- prologue: async-stage chunk 0 into buffer 0 (LDS written directly) --
  ASYNC_CP_B128(gA0, &As[0][so0]);
  ASYNC_CP_B128(gA1, &As[0][so1]);
  ASYNC_CP_B128(gB0, &Bs[0][so0]);
  ASYNC_CP_B128(gB1, &Bs[0][so1]);
  gA0 += 32; gA1 += 32; gB0 += 32; gB1 += 32;
#else
  v8h ra0 = *(const v8h*)gA0;  v8h ra1 = *(const v8h*)gA1;
  v8h rb0 = *(const v8h*)gB0;  v8h rb1 = *(const v8h*)gB1;
  gA0 += 32; gA1 += 32; gB0 += 32; gB1 += 32;
  *(v8h*)&As[0][so0] = ra0;  *(v8h*)&As[0][so1] = ra1;
  *(v8h*)&Bs[0][so0] = rb0;  *(v8h*)&Bs[0][so1] = rb1;
#endif

  const int bkA = half_id * 8;         // A frag: K {0..7,16..23} / {8..15,24..31}
  const int bkB = half_id * 8;         // B frag: K {0..7,16..23} / {8..15,24..31}
  const int aRow = wm * 32 + lrow;
  const int bRow = wn * 64 + lrow;

  for (int k = 0; k < nk; ++k) {
    const int cur = k & 1;
    const int nxt = cur ^ 1;
    const bool more = (k + 1) < nk;

    wait_async_zero();                 // my LDS writes for buf[cur] complete
    __syncthreads();                   // everyone's writes visible

    // ---- issue staging for chunk k+1 while computing chunk k ----
#if ASYNC_LDS
    if (more) {
      ASYNC_CP_B128(gA0, &As[nxt][so0]);
      ASYNC_CP_B128(gA1, &As[nxt][so1]);
      ASYNC_CP_B128(gB0, &Bs[nxt][so0]);
      ASYNC_CP_B128(gB1, &Bs[nxt][so1]);
      gA0 += 32; gA1 += 32; gB0 += 32; gB1 += 32;
    }
#else
    v8h ta0, ta1, tb0, tb1;
    if (more) {
      ta0 = *(const v8h*)gA0;  ta1 = *(const v8h*)gA1;
      tb0 = *(const v8h*)gB0;  tb1 = *(const v8h*)gB1;
      gA0 += 32; gA1 += 32; gB0 += 32; gB1 += 32;
    }
#endif

    // ---- fragments from LDS buf[cur] ----
    const _Float16* as = As[cur];
    const _Float16* bs = Bs[cur];
    v16h a[2], b[4];
#pragma unroll
    for (int mi = 0; mi < 2; ++mi) {
      v8h lo = *(const v8h*)&as[(aRow + mi * 16) * LDSS + bkA];
      v8h hi = *(const v8h*)&as[(aRow + mi * 16) * LDSS + bkA + 16];
      a[mi] = __builtin_shufflevector(lo, hi, 0,1,2,3,4,5,6,7,8,9,10,11,12,13,14,15);
    }
#pragma unroll
    for (int ni = 0; ni < 4; ++ni) {
      v8h lo = *(const v8h*)&bs[(bRow + ni * 16) * LDSS + bkB];
      v8h hi = *(const v8h*)&bs[(bRow + ni * 16) * LDSS + bkB + 16];
      b[ni] = __builtin_shufflevector(lo, hi, 0,1,2,3,4,5,6,7,8,9,10,11,12,13,14,15);
    }

    // ---- 8 WMMA per K-chunk ----
#pragma unroll
    for (int mi = 0; mi < 2; ++mi)
#pragma unroll
      for (int ni = 0; ni < 4; ++ni)
        acc[mi][ni] = __builtin_amdgcn_wmma_f32_16x16x32_f16(
            false, a[mi], false, b[ni], (short)0, acc[mi][ni], false, false);

#if !ASYNC_LDS
    if (more) {
      *(v8h*)&As[nxt][so0] = ta0;  *(v8h*)&As[nxt][so1] = ta1;
      *(v8h*)&Bs[nxt][so0] = tb0;  *(v8h*)&Bs[nxt][so1] = tb1;
    }
#endif
  }

  // ---- store C (16x16 f32 C/D layout: VGPR r -> M=r / r+8) ----
#pragma unroll
  for (int mi = 0; mi < 2; ++mi) {
    int rbase = m0 + wm * 32 + mi * 16 + half_id * 8;
#pragma unroll
    for (int ni = 0; ni < 4; ++ni) {
      int col = n0 + wn * 64 + ni * 16 + lrow;
#pragma unroll
      for (int r = 0; r < 8; ++r)
        C[(size_t)(rbase + r) * N + col] = acc[mi][ni][r];
    }
  }
}

// ---------------------------------------------------------------------------
// LIF scans: one thread per (batch, neuron), coalesced streaming over T.
// ---------------------------------------------------------------------------
__global__ void lif_scan_tohalf(const float* __restrict__ Z, _Float16* __restrict__ S,
                                int stride, int T) {
  int idx = blockIdx.x * blockDim.x + threadIdx.x;
  if (idx >= stride) return;
  S[idx] = (_Float16)0.0f;                   // t = 0 block
  float I = 0.0f, V = 0.0f;
  for (int t = 1; t < T; ++t) {
    if (t + 1 < T) __builtin_prefetch(&Z[(size_t)t * stride + idx], 0, 1);
    float z  = Z[(size_t)(t - 1) * stride + idx];
    I = ALPHA_F * I + z;
    float Vp = BETA_F * V + (1.0f - BETA_F) * I;
    float s  = (Vp > 1.0f) ? 1.0f : 0.0f;
    V = (1.0f - s) * Vp;
    if (t < T - 1) S[(size_t)t * stride + idx] = (_Float16)s;
  }
}

__global__ void lif_scan_out(const float* __restrict__ Z, float* __restrict__ out,
                             int stride, int T) {
  int idx = blockIdx.x * blockDim.x + threadIdx.x;
  if (idx >= stride) return;
  out[idx] = 0.0f;                           // t = 0 block
  float I = 0.0f, V = 0.0f;
  for (int t = 1; t < T; ++t) {
    if (t + 1 < T) __builtin_prefetch(&Z[(size_t)t * stride + idx], 0, 1);
    float z  = Z[(size_t)(t - 1) * stride + idx];
    I = ALPHA_F * I + z;
    float Vp = BETA_F * V + (1.0f - BETA_F) * I;
    float s  = (Vp > 1.0f) ? 1.0f : 0.0f;
    V = (1.0f - s) * Vp;
    out[(size_t)t * stride + idx] = s;
  }
}

// ---------------------------------------------------------------------------
extern "C" void kernel_launch(void* const* d_in, const int* in_sizes, int n_in,
                              void* d_out, int out_size, void* d_ws, size_t ws_size,
                              hipStream_t stream) {
  const float* x  = (const float*)d_in[0];   // [100, 256, 784]
  const float* W1 = (const float*)d_in[1];   // [1024, 784]
  const float* W2 = (const float*)d_in[2];   // [256, 1024]
  float* out = (float*)d_out;                // [100, 256, 256]

  const int M = (T_STEPS - 1) * BATCH;       // 25344 time-parallel GEMM rows

  // Workspace carve-out (256B aligned). Z2 aliases Xh (dead after GEMM1).
  char* ws = (char*)d_ws;
  size_t o = 0;
  auto nxt = [&](size_t bytes) { size_t p = o; o += (bytes + 255) & ~(size_t)255; return p; };
  _Float16* Xh  = (_Float16*)(ws + nxt((size_t)M  * K1PAD * sizeof(_Float16)));  // 40.5 MB
  _Float16* W1h = (_Float16*)(ws + nxt((size_t)D1 * K1PAD * sizeof(_Float16)));  //  1.6 MB
  _Float16* W2h = (_Float16*)(ws + nxt((size_t)D2 * D1    * sizeof(_Float16)));  //  0.5 MB
  float*    Z1  = (float*)   (ws + nxt((size_t)M  * D1    * sizeof(float)));     // 104  MB
  _Float16* S1h = (_Float16*)(ws + nxt((size_t)M  * D1    * sizeof(_Float16)));  //  52  MB
  float*    Z2  = (float*)Xh;                                                    // alias (26 MB)

  const int THR = 256;

  // Stage inputs to f16 (pad K: 784 -> 800 so the WMMA K-loop is exact).
  { int n = M * K1PAD;
    cvt_pad_kernel<<<(n + THR - 1) / THR, THR, 0, stream>>>(x,  Xh,  M,  D0, K1PAD); }
  { int n = D1 * K1PAD;
    cvt_pad_kernel<<<(n + THR - 1) / THR, THR, 0, stream>>>(W1, W1h, D1, D0, K1PAD); }
  { int n = D2 * D1;
    cvt_pad_kernel<<<(n + THR - 1) / THR, THR, 0, stream>>>(W2, W2h, D2, D1, D1); }

  // Layer 1: time-parallel GEMM then sequential elementwise LIF scan.
  gemm_wmma_f16<<<dim3(M / 128, D1 / 128), THR, 0, stream>>>(Xh, W1h, Z1, M, D1, K1PAD);
  lif_scan_tohalf<<<(BATCH * D1 + THR - 1) / THR, THR, 0, stream>>>(Z1, S1h, BATCH * D1, T_STEPS);

  // Layer 2.
  gemm_wmma_f16<<<dim3(M / 128, D2 / 128), THR, 0, stream>>>(S1h, W2h, Z2, M, D2, D1);
  lif_scan_out<<<(BATCH * D2 + THR - 1) / THR, THR, 0, stream>>>(Z2, out, BATCH * D2, T_STEPS);
}
